// FusedLayernormMMReplacement_25623774888305
// MI455X (gfx1250) — matrix operand order repaired and checked by reading, not compiled, not run on hardware
//
#include <hip/hip_runtime.h>
#include <hip/hip_bf16.h>

// ---------------------------------------------------------------------------
// Fused LayerNorm + QKV projection for MI455X (gfx1250, wave32, WMMA + TDM).
//   M = B*S = 8192, K = D = 2048, N = 2048 per projection (Q,K,V), out [M,6144]
// Pipeline:
//   1) layernorm rows (fp32 math) -> bf16 activations in d_ws
//   2) cast 3 weight matrices fp32 -> bf16 in d_ws
//   3) WMMA GEMM: tensor_load_to_lds (TDM, double buffered, padded LDS rows)
//      + ds_load fragments + v_wmma_f32_16x16x32_bf16, bias in fp32 epilogue
// ---------------------------------------------------------------------------

typedef __attribute__((ext_vector_type(16))) __bf16 v16bf;
typedef __attribute__((ext_vector_type(8)))  __bf16 v8bf;
typedef __attribute__((ext_vector_type(8)))  float  v8f;
typedef __attribute__((ext_vector_type(4)))  unsigned int v4u;
typedef __attribute__((ext_vector_type(8)))  int v8i;
typedef __attribute__((ext_vector_type(4)))  int v4i;

#define M_TOT 8192
#define K_DIM 2048
#define N_PER 2048
#define N_TOT 6144
#define EPSF  1e-5f

#define KSTEP       64                 // K elems staged per buffer
#define LDS_STRIDE  80                 // halves per LDS tile row (64 + 16 pad) = 160B
#define A_ROWS      128
#define B_ROWS      256
#define A_TILE_B    (A_ROWS * LDS_STRIDE * 2)   // 20480 B
#define B_TILE_B    (B_ROWS * LDS_STRIDE * 2)   // 40960 B
#define LDS_BYTES   (2 * (A_TILE_B + B_TILE_B)) // 122880 B (double buffered)

__device__ __forceinline__ unsigned short f2bf_rne(float f) {
    unsigned u = __float_as_uint(f);
    unsigned r = u + 0x7FFFu + ((u >> 16) & 1u);   // round-to-nearest-even
    return (unsigned short)(r >> 16);
}

// ---- kernel 1: LayerNorm one row per block, 256 threads, 8 elems/thread ----
__global__ __launch_bounds__(256)
void ln_bf16_kernel(const float* __restrict__ x,
                    const float* __restrict__ gamma,
                    const float* __restrict__ beta,
                    __bf16* __restrict__ y) {
    __shared__ float ssum[256];
    __shared__ float ssq[256];
    const int row = blockIdx.x;
    const int t   = threadIdx.x;
    const float4* xr = (const float4*)(x + (size_t)row * K_DIM);

    float4 v0 = xr[t * 2 + 0];
    float4 v1 = xr[t * 2 + 1];
    float e[8] = {v0.x, v0.y, v0.z, v0.w, v1.x, v1.y, v1.z, v1.w};

    float s = 0.f, q = 0.f;
#pragma unroll
    for (int j = 0; j < 8; ++j) { s += e[j]; q += e[j] * e[j]; }
    ssum[t] = s; ssq[t] = q;
    __syncthreads();
#pragma unroll
    for (int stride = 128; stride > 0; stride >>= 1) {
        if (t < stride) { ssum[t] += ssum[t + stride]; ssq[t] += ssq[t + stride]; }
        __syncthreads();
    }
    const float mu  = ssum[0] * (1.0f / K_DIM);
    const float var = ssq[0] * (1.0f / K_DIM) - mu * mu;
    const float rs  = rsqrtf(var + EPSF);

    const int base = t * 8;
    unsigned short o[8];
#pragma unroll
    for (int j = 0; j < 8; ++j) {
        float f = (e[j] - mu) * rs * gamma[base + j] + beta[base + j];
        o[j] = f2bf_rne(f);
    }
    uint4 packed;
    packed.x = (unsigned)o[0] | ((unsigned)o[1] << 16);
    packed.y = (unsigned)o[2] | ((unsigned)o[3] << 16);
    packed.z = (unsigned)o[4] | ((unsigned)o[5] << 16);
    packed.w = (unsigned)o[6] | ((unsigned)o[7] << 16);
    *(uint4*)((unsigned short*)y + (size_t)row * K_DIM + base) = packed;
}

// ---- kernel 2: fp32 -> bf16 cast of the 3 weight matrices -----------------
__global__ __launch_bounds__(256)
void cast_w_kernel(const float* __restrict__ qw,
                   const float* __restrict__ kw,
                   const float* __restrict__ vw,
                   unsigned short* __restrict__ dst) {
    const size_t per = (size_t)N_PER * K_DIM / 4;      // float4 quads per matrix
    size_t i = (size_t)blockIdx.x * blockDim.x + threadIdx.x;
    if (i >= 3 * per) return;
    const size_t mat = i / per;
    const float* src = (mat == 0) ? qw : (mat == 1) ? kw : vw;
    const size_t off = (i - mat * per) * 4;
    float4 f = *(const float4*)(src + off);
    ushort4 o;
    o.x = f2bf_rne(f.x); o.y = f2bf_rne(f.y);
    o.z = f2bf_rne(f.z); o.w = f2bf_rne(f.w);
    *(ushort4*)(dst + mat * (size_t)N_PER * K_DIM + off) = o;
}

// ---- TDM: issue one 2D tile load (rows x 64 bf16) into padded LDS ---------
// D# per CDNA5 ISA: group0 = {flags, lds_addr,
// global_addr[56:0], type=2}; group1 = {data_size=2B, pad 32B per 128B row,
// tensor_dim0=K_DIM, tile_dim0=64, tile_dim1=rows, tensor_dim0_stride=K_DIM}.
__device__ __forceinline__ void tdm_load_tile(const __bf16* gsrc,
                                              unsigned lds_byte_addr,
                                              unsigned tile_rows) {
    unsigned long long ga = (unsigned long long)(const void*)gsrc;
    v4u g0;
    g0[0] = 1u;                                    // count=1 user descriptor
    g0[1] = lds_byte_addr;                         // lds_addr (bytes)
    g0[2] = (unsigned)(ga & 0xFFFFFFFFu);          // global_addr[31:0]
    g0[3] = (unsigned)((ga >> 32) & 0x01FFFFFFu)   // global_addr[56:32]
          | (2u << 30);                            // type = 2 ("image")
    v8i g1;
    g1[0] = (int)((1u << 16)                       // data_size = 2 bytes
                | (1u << 20)                       // pad_enable
                | (4u << 22)                       // pad_interval = 32 DWORDs
                | (7u << 25));                     // pad_amount  = 8 DWORDs (32B)
    g1[1] = (int)(((unsigned)K_DIM & 0xFFFFu) << 16);   // tensor_dim0 lo16 @ bits 63:48
    g1[2] = (int)(((unsigned)K_DIM >> 16)               // tensor_dim0 hi16
                | (0xFFFFu << 16));                     // tensor_dim1 lo16 (large)
    g1[3] = (int)(0x7FFFu                               // tensor_dim1 hi16
                | ((unsigned)KSTEP << 16));             // tile_dim0 = 64
    g1[4] = (int)(tile_rows & 0xFFFFu);                 // tile_dim1; tile_dim2 = 0
    g1[5] = (int)(unsigned)K_DIM;                       // tensor_dim0_stride lo32
    g1[6] = 0;                                          // stride hi16 | dim1_stride lo
    g1[7] = 0;
    v4i z4 = {0, 0, 0, 0};
#if defined(__clang_major__) && (__clang_major__ >= 23)
    v8i z8 = {0, 0, 0, 0, 0, 0, 0, 0};
    __builtin_amdgcn_tensor_load_to_lds(g0, g1, z4, z4, z8, 0);
#else
    __builtin_amdgcn_tensor_load_to_lds(g0, g1, z4, z4, 0);
#endif
}

// ---- kernel 3: WMMA GEMM with TDM-staged LDS tiles ------------------------
// block = 256 threads = 8 waves (2 in M x 4 in N); block tile 128x256,
// wave tile 64x64 = 4x4 v_wmma_f32_16x16x32_bf16; KSTEP=64 (2 substeps),
// double-buffered TDM loads synchronized via s_wait_tensorcnt + barriers.
__global__ __launch_bounds__(256)
void qkv_wmma_kernel(const __bf16* __restrict__ A,   // [M_TOT, K_DIM] bf16
                     const __bf16* __restrict__ W,   // [3, N_PER, K_DIM] bf16
                     const float* __restrict__ qb,
                     const float* __restrict__ kb,
                     const float* __restrict__ vb,
                     float* __restrict__ out) {      // [M_TOT, N_TOT] fp32
    extern __shared__ __bf16 lds_dyn[];
    const unsigned lds_base = (unsigned)(size_t)(const void*)lds_dyn;

    const int lane = threadIdx.x & 31;
    const int wave = threadIdx.x >> 5;
    const int wm = wave >> 2;           // 0..1
    const int wn = wave & 3;            // 0..3
    const int m0b = blockIdx.y * 128;                // block M origin
    const int n0b = blockIdx.x * 256;                // block N origin (global)
    const int m0  = m0b + wm * 64;
    const int n0g = n0b + wn * 64;
    const int proj = n0b / N_PER;                    // block never straddles
    const int ncol = n0g - proj * N_PER;
    const __bf16* Wp = W + (size_t)proj * N_PER * K_DIM + (size_t)(n0b - proj * N_PER) * K_DIM;
    const int lhi = lane >> 4;          // lane group 0/1
    const int llo = lane & 15;

    // LDS partition: [A0][A1][B0][B1]
    __bf16* ldsAg = lds_dyn;                                     // generic ptrs
    __bf16* ldsBg = lds_dyn + 2 * (A_TILE_B / 2);
    const unsigned ldsA_addr = lds_base;
    const unsigned ldsB_addr = lds_base + 2 * A_TILE_B;

    v8f acc[4][4];
#pragma unroll
    for (int mi = 0; mi < 4; ++mi)
#pragma unroll
        for (int ni = 0; ni < 4; ++ni)
            acc[mi][ni] = (v8f){0.f, 0.f, 0.f, 0.f, 0.f, 0.f, 0.f, 0.f};

    // prologue: stage k-tile 0 into buffer 0 (wave0 -> A, wave1 -> B)
    if (wave == 0) {
        tdm_load_tile(A + (size_t)m0b * K_DIM, ldsA_addr, A_ROWS);
    } else if (wave == 1) {
        tdm_load_tile(Wp, ldsB_addr, B_ROWS);
    }

    const int KT = K_DIM / KSTEP;       // 32 k-tiles
    for (int kt = 0; kt < KT; ++kt) {
        const int nb = kt + 1;
        if (nb < KT) {                  // stage next k-tile into other buffer
            const int b = nb & 1;
            if (wave == 0) {
                tdm_load_tile(A + (size_t)m0b * K_DIM + nb * KSTEP,
                              ldsA_addr + b * A_TILE_B, A_ROWS);
            } else if (wave == 1) {
                tdm_load_tile(Wp + nb * KSTEP,
                              ldsB_addr + b * B_TILE_B, B_ROWS);
            }
            if (wave < 2) {
                // two TDM ops in flight; oldest (current buffer) must be done
                __builtin_amdgcn_s_wait_tensorcnt(1);
            }
        } else {
            if (wave < 2) {
                // last tile: drain all outstanding TDM ops
                __builtin_amdgcn_s_wait_tensorcnt(0);
            }
        }
        __syncthreads();                // tile visible to whole workgroup

        const int b = kt & 1;
        const __bf16* At = ldsAg + b * (A_TILE_B / 2) + wm * 64 * LDS_STRIDE;
        const __bf16* Bt = ldsBg + b * (B_TILE_B / 2) + wn * 64 * LDS_STRIDE;
#pragma unroll
        for (int sub = 0; sub < KSTEP; sub += 32) {
            v16bf afrag[4];
            v16bf bfrag[4];
            // A frag (16x32): lane M=llo; halves 0-7 @K=sub+8*lhi, 8-15 @+16
#pragma unroll
            for (int mi = 0; mi < 4; ++mi) {
                const __bf16* p = At + (mi * 16 + llo) * LDS_STRIDE + sub + 8 * lhi;
                v8bf lo = *(const v8bf*)p;
                v8bf hi = *(const v8bf*)(p + 16);
                afrag[mi] = __builtin_shufflevector(lo, hi,
                    0, 1, 2, 3, 4, 5, 6, 7, 8, 9, 10, 11, 12, 13, 14, 15);
            }
            // B frag (32x16): lane N=llo; halves 0-7 @K=sub+8*lhi, 8-15 @+16
#pragma unroll
            for (int ni = 0; ni < 4; ++ni) {
                const __bf16* p = Bt + (ni * 16 + llo) * LDS_STRIDE + sub + 8 * lhi;
                v8bf lo = *(const v8bf*)p;
                v8bf hi = *(const v8bf*)(p + 16);
                bfrag[ni] = __builtin_shufflevector(lo, hi,
                    0, 1, 2, 3, 4, 5, 6, 7, 8, 9, 10, 11, 12, 13, 14, 15);
            }
#pragma unroll
            for (int mi = 0; mi < 4; ++mi)
#pragma unroll
                for (int ni = 0; ni < 4; ++ni)
                    acc[mi][ni] = __builtin_amdgcn_wmma_f32_16x16x32_bf16(
                        false, afrag[mi], false, bfrag[ni],
                        (short)0, acc[mi][ni], false, false);
        }
        __syncthreads();                // done reading before buffer reuse
    }

    const float* bias = (proj == 0) ? qb : (proj == 1) ? kb : vb;
#pragma unroll
    for (int ni = 0; ni < 4; ++ni) {
        const float bv = bias[ncol + ni * 16 + llo];
        const int gn = n0g + ni * 16 + llo;
#pragma unroll
        for (int mi = 0; mi < 4; ++mi) {
#pragma unroll
            for (int r = 0; r < 8; ++r) {
                const int gm = m0 + mi * 16 + r + 8 * lhi;   // C/D layout
                out[(size_t)gm * N_TOT + gn] = acc[mi][ni][r] + bv;
            }
        }
    }
}

// ---------------------------------------------------------------------------
extern "C" void kernel_launch(void* const* d_in, const int* in_sizes, int n_in,
                              void* d_out, int out_size, void* d_ws, size_t ws_size,
                              hipStream_t stream) {
    const float* x  = (const float*)d_in[0];
    const float* nw = (const float*)d_in[1];
    const float* nb = (const float*)d_in[2];
    const float* qw = (const float*)d_in[3];
    const float* qb = (const float*)d_in[4];
    const float* kw = (const float*)d_in[5];
    const float* kb = (const float*)d_in[6];
    const float* vw = (const float*)d_in[7];
    const float* vb = (const float*)d_in[8];

    __bf16* Abf = (__bf16*)d_ws;                                   // 32 MB
    unsigned short* Wbf = (unsigned short*)((char*)d_ws +
                          (size_t)M_TOT * K_DIM * sizeof(__bf16)); // 24 MB

    ln_bf16_kernel<<<M_TOT, 256, 0, stream>>>(x, nw, nb, Abf);

    const size_t quads = 3ull * N_PER * K_DIM / 4;
    cast_w_kernel<<<(unsigned)((quads + 255) / 256), 256, 0, stream>>>(qw, kw, vw, Wbf);

    dim3 grid(N_TOT / 256, M_TOT / 128);
    qkv_wmma_kernel<<<grid, 256, LDS_BYTES, stream>>>(Abf, (const __bf16*)Wbf,
                                                      qb, kb, vb, (float*)d_out);
}
